// MaskDecoderLayer_64209761075348
// MI455X (gfx1250) — matrix operand order repaired and checked by reading, not compiled, not run on hardware
//
#include <hip/hip_runtime.h>
#include <hip/hip_bf16.h>
#include <cstdint>

typedef __attribute__((ext_vector_type(16))) __bf16 v16bf;
typedef __attribute__((ext_vector_type(8)))  float  v8f;

// Problem constants (from reference)
static constexpr int Bc  = 16;
static constexpr int Mc  = 128;
static constexpr int Lc  = 2048;
static constexpr int Ec  = 1024;
static constexpr int Hc  = 16;
static constexpr int DHc = 64;
static constexpr int HIDc = 2048;

#if __has_builtin(__builtin_amdgcn_tensor_load_to_lds)
#define HAVE_TDM 1
#else
#define HAVE_TDM 0
#endif

// ------------------------------------------------------------ TDM helpers
#if HAVE_TDM
typedef unsigned int v4u  __attribute__((ext_vector_type(4)));
typedef int          v8i_ __attribute__((ext_vector_type(8)));
typedef int          v4i_ __attribute__((ext_vector_type(4)));

// Issue a 2D tile DMA: tileD1 rows of tileD0 elements (2-byte), row stride
// rowStrideElems, into LDS at ldsAddr. Optional LDS row padding via the D#
// pad feature (padIntervalCode: 0=2 dwords .. 5=64; padAmountCode: 0=1 .. 7=8).
__device__ __forceinline__ void tdm_load_2d(unsigned ldsAddr, const void* gaddr,
                                            unsigned tileD0, unsigned tileD1,
                                            unsigned rowStrideElems,
                                            unsigned padIntervalCode,
                                            unsigned padAmountCode, bool padEn)
{
  const unsigned long long ga = (unsigned long long)(uintptr_t)gaddr;
  v4u g0;
  g0[0] = 1u;                                                   // count=1 (valid)
  g0[1] = ldsAddr;                                              // lds_addr [63:32]
  g0[2] = (unsigned)(ga & 0xFFFFFFFFu);                         // global_addr lo
  g0[3] = (unsigned)((ga >> 32) & 0x01FFFFFFu) | (2u << 30);    // global_addr hi | type=2

  unsigned w0 = (1u << 16);                                     // data_size=1 (2 bytes)
  if (padEn) w0 |= (1u << 20) | (padIntervalCode << 22) | (padAmountCode << 25);
  v8i_ g1;
  g1[0] = (int)w0;
  g1[1] = (int)((rowStrideElems & 0xFFFFu) << 16);              // tensor_dim0 lo16 (abar=0)
  g1[2] = (int)((rowStrideElems >> 16) | ((tileD1 & 0xFFFFu) << 16)); // dim0 hi16 | tensor_dim1 lo16
  g1[3] = (int)((tileD1 >> 16) | (tileD0 << 16));               // tensor_dim1 hi16 | tile_dim0
  g1[4] = (int)(tileD1 & 0xFFFFu);                              // tile_dim1 | tile_dim2=0
  g1[5] = (int)rowStrideElems;                                  // tensor_dim0_stride lo32
  g1[6] = 0;                                                    // stride hi16 | dim1_stride lo16
  g1[7] = 0;

  v4i_ gz = {0, 0, 0, 0};
#if __clang_major__ >= 23
  v8i_ gz8 = {0, 0, 0, 0, 0, 0, 0, 0};
  __builtin_amdgcn_tensor_load_to_lds(g0, g1, gz, gz, gz8, 0);
#else
  __builtin_amdgcn_tensor_load_to_lds(g0, g1, gz, gz, 0);
#endif
}
#endif  // HAVE_TDM

__device__ __forceinline__ void tdm_wait0()
{
#if __has_builtin(__builtin_amdgcn_s_wait_tensorcnt)
  __builtin_amdgcn_s_wait_tensorcnt(0);
#else
  asm volatile("s_wait_tensorcnt 0x0" ::: "memory");
#endif
}
__device__ __forceinline__ void tdm_wait2()
{
#if __has_builtin(__builtin_amdgcn_s_wait_tensorcnt)
  __builtin_amdgcn_s_wait_tensorcnt(2);
#else
  asm volatile("s_wait_tensorcnt 0x2" ::: "memory");
#endif
}

// ---------------------------------------------------------------- convert
__global__ void k_f32_to_bf16(const float* __restrict__ in, __bf16* __restrict__ out, long n) {
  long i = (long)blockIdx.x * blockDim.x + threadIdx.x;
  if (i < n) out[i] = (__bf16)in[i];
}

// ---------------------------------------------------------------- GEMM
// C[rows x N] = A_bf16[rows x K] @ W_bf16[K x N] + bias (+ resid), stores
// compile-time selected f32/bf16. 128x128 block tile, 8 waves, each wave a
// 32x64 tile as 2x4 WMMA 16x16x32 bf16 fragments. Double-buffered TDM tile
// pipeline when available (wave 0 issues stage i+1 before waiting stage i).
template <bool RES, bool OF, bool OB>
__global__ __launch_bounds__(256) void k_gemm_bf16(
    const __bf16* __restrict__ A, const __bf16* __restrict__ W,
    const float* __restrict__ bias, const float* __restrict__ resid,
    float* __restrict__ outF, __bf16* __restrict__ outB,
    int rows, int N, int K)
{
#if HAVE_TDM
  constexpr int NBUF = 2;
#else
  constexpr int NBUF = 1;
#endif
  __shared__ __bf16 sA[NBUF][128][32];
  __shared__ __bf16 sB[NBUF][32][144];   // 128 data + 16 pad per row

  const int tid  = threadIdx.x;
  const int lane = tid & 31;
  const int wid  = tid >> 5;
  const int wm   = wid >> 1;   // 0..3 -> 32 rows each
  const int wn   = wid & 1;    // 0..1 -> 64 cols each
  const int ln   = lane & 15;
  const int hi   = lane >> 4;

  const int rowBase = blockIdx.y * 128;
  const int colBase = blockIdx.x * 128;

  v8f zacc = {};
  v8f acc[2][4];
  #pragma unroll
  for (int mt = 0; mt < 2; mt++)
    #pragma unroll
    for (int nt = 0; nt < 4; nt++) acc[mt][nt] = zacc;

#if HAVE_TDM
  // TDM ignores EXEC; gate issue with a uniform branch on wave id.
  auto issueStage = [&](int bi, int kb2) {
    tdm_load_2d((unsigned)(uintptr_t)&sA[bi][0][0],
                A + (size_t)rowBase * K + kb2,
                /*tileD0=*/32, /*tileD1=*/128, /*stride=*/(unsigned)K,
                0, 0, /*pad=*/false);
    tdm_load_2d((unsigned)(uintptr_t)&sB[bi][0][0],
                W + (size_t)kb2 * N + colBase,
                /*tileD0=*/128, /*tileD1=*/32, /*stride=*/(unsigned)N,
                /*interval 64 dwords=*/5, /*amount 8 dwords=*/7, /*pad=*/true);
  };
  if (wid == 0) issueStage(0, 0);
#else
  const int arow = tid >> 1;         // 128 rows, 2 threads/row
  const int acol = (tid & 1) * 16;
  const int brow = tid >> 3;         // 32 rows, 8 threads/row
  const int bcol = (tid & 7) * 16;
#endif

  int buf = 0;
  for (int kb = 0; kb < K; kb += 32) {
#if HAVE_TDM
    if (wid == 0) {
      if (kb + 32 < K) {
        issueStage(buf ^ 1, kb + 32);   // overlap next stage's DMA with this compute
        tdm_wait2();                     // newest 2 descriptors may remain in flight
      } else {
        tdm_wait0();
      }
    }
    __syncthreads();
#else
    {
      const uint4* src = reinterpret_cast<const uint4*>(A + (size_t)(rowBase + arow) * K + kb + acol);
      uint4 u0 = src[0], u1 = src[1];
      *reinterpret_cast<uint4*>(&sA[0][arow][acol])     = u0;
      *reinterpret_cast<uint4*>(&sA[0][arow][acol + 8]) = u1;
    }
    {
      const uint4* src = reinterpret_cast<const uint4*>(W + (size_t)(kb + brow) * N + colBase + bcol);
      uint4 u0 = src[0], u1 = src[1];
      *reinterpret_cast<uint4*>(&sB[0][brow][bcol])     = u0;
      *reinterpret_cast<uint4*>(&sB[0][brow][bcol + 8]) = u1;
    }
    __syncthreads();
#endif

    v16bf af[2];
    #pragma unroll
    for (int mt = 0; mt < 2; mt++) {
      const int r = wm * 32 + mt * 16 + ln;
      #pragma unroll
      for (int i = 0; i < 8; i++) {
        const int kk = (i < 4 ? 0 : 16) + hi * 8 + (i & 3) * 2;
        af[mt][2 * i]     = sA[buf][r][kk];
        af[mt][2 * i + 1] = sA[buf][r][kk + 1];
      }
    }
    v16bf bfm[4];
    #pragma unroll
    for (int nt = 0; nt < 4; nt++) {
      const int c = wn * 64 + nt * 16 + ln;
      #pragma unroll
      for (int i = 0; i < 8; i++) {
        const int kk = (i < 4 ? 0 : 16) + hi * 8 + (i & 3) * 2;
        bfm[nt][2 * i]     = sB[buf][kk][c];
        bfm[nt][2 * i + 1] = sB[buf][kk + 1][c];
      }
    }
    #pragma unroll
    for (int mt = 0; mt < 2; mt++)
      #pragma unroll
      for (int nt = 0; nt < 4; nt++)
        acc[mt][nt] = __builtin_amdgcn_wmma_f32_16x16x32_bf16(
            false, af[mt], false, bfm[nt], (short)0, acc[mt][nt], false, false);
    __syncthreads();
    buf ^= (NBUF - 1);
  }

  #pragma unroll
  for (int mt = 0; mt < 2; mt++) {
    #pragma unroll
    for (int nt = 0; nt < 4; nt++) {
      const int col = colBase + wn * 64 + nt * 16 + ln;
      const float bv = bias[col];
      #pragma unroll
      for (int r = 0; r < 8; r++) {
        const int row = rowBase + wm * 32 + mt * 16 + hi * 8 + r;
        float v = acc[mt][nt][r] + bv;
        const size_t idx = (size_t)row * N + col;
        if constexpr (RES) v += resid[idx];
        if constexpr (OF)  outF[idx] = v;
        if constexpr (OB)  outB[idx] = (__bf16)v;
      }
    }
  }
}

// ---------------------------------------------------------------- flash attention
// One wave per (b, h, 16-row q tile). Online softmax over 32-wide KV chunks.
// amq (may be null): fully-resolved attend byte [B, Qlen, Klen].
__global__ __launch_bounds__(32) void k_attn(
    const __bf16* __restrict__ Qp, const __bf16* __restrict__ Kp,
    const __bf16* __restrict__ Vp, __bf16* __restrict__ Op,
    const unsigned char* __restrict__ amq,
    int Qlen, int Klen, long long qStride, long long kStride)
{
  __shared__ __bf16 sK[32][80];   // 64 data + 16 pad per row (160B, 16B aligned)
  __shared__ __bf16 sV[32][80];
  __shared__ __bf16 sP[16][40];

  const int lane = threadIdx.x;
  const int ln = lane & 15, hi = lane >> 4;
  const int b = blockIdx.z, h = blockIdx.y, qt = blockIdx.x;

  const __bf16* Qb = Qp + (size_t)b * qStride + (size_t)h * DHc;
  const __bf16* Kb = Kp + (size_t)b * kStride + (size_t)h * DHc;
  const __bf16* Vb = Vp + (size_t)b * kStride + (size_t)h * DHc;
  __bf16*       Ob = Op + (size_t)b * qStride + (size_t)h * DHc;

  // Q fragments (16x64 as two 16x32 A-frags), loaded once
  v16bf aq[2];
  {
    const __bf16* qrow = Qb + (size_t)(qt * 16 + ln) * Ec;
    #pragma unroll
    for (int f = 0; f < 2; f++)
      #pragma unroll
      for (int i = 0; i < 8; i++) {
        const int k = f * 32 + (i < 4 ? 0 : 16) + hi * 8 + (i & 3) * 2;
        aq[f][2 * i]     = qrow[k];
        aq[f][2 * i + 1] = qrow[k + 1];
      }
  }

  float rowMax[8], rowSum[8];
  #pragma unroll
  for (int r = 0; r < 8; r++) { rowMax[r] = -3.0e38f; rowSum[r] = 0.0f; }
  v8f z = {};
  v8f oacc[4];
  #pragma unroll
  for (int c = 0; c < 4; c++) oacc[c] = z;

  const float scl = 0.125f;     // 1/sqrt(DH=64)
  const int nChunks = Klen / 32;
  for (int j = 0; j < nChunks; j++) {
    const int kvBase = j * 32;
#if HAVE_TDM
    // 32 rows x 64 elems, LDS rows padded 64->80 elems:
    // 32 dwords data (interval code 4) + 8 dwords pad (amount code 7)
    tdm_load_2d((unsigned)(uintptr_t)&sK[0][0], Kb + (size_t)kvBase * Ec,
                64, 32, (unsigned)Ec, 4, 7, true);
    tdm_load_2d((unsigned)(uintptr_t)&sV[0][0], Vb + (size_t)kvBase * Ec,
                64, 32, (unsigned)Ec, 4, 7, true);
    tdm_wait0();
#else
    {
      const __bf16* kr = Kb + (size_t)(kvBase + lane) * Ec;
      const __bf16* vr = Vb + (size_t)(kvBase + lane) * Ec;
      #pragma unroll
      for (int u = 0; u < 8; u++) {
        *reinterpret_cast<uint4*>(&sK[lane][u * 8]) = reinterpret_cast<const uint4*>(kr)[u];
        *reinterpret_cast<uint4*>(&sV[lane][u * 8]) = reinterpret_cast<const uint4*>(vr)[u];
      }
    }
#endif
    __syncthreads();

    // S(16x32) = Q(16x64) * K^T(64x32): 2 col-groups x 2 K-halves
    v8f s[2]; s[0] = z; s[1] = z;
    #pragma unroll
    for (int g = 0; g < 2; g++) {
      #pragma unroll
      for (int f = 0; f < 2; f++) {
        v16bf bk;
        #pragma unroll
        for (int i = 0; i < 8; i++) {
          const int e = f * 32 + (i < 4 ? 0 : 16) + hi * 8 + (i & 3) * 2;
          bk[2 * i]     = sK[g * 16 + ln][e];
          bk[2 * i + 1] = sK[g * 16 + ln][e + 1];
        }
        s[g] = __builtin_amdgcn_wmma_f32_16x16x32_bf16(false, aq[f], false, bk,
                                                       (short)0, s[g], false, false);
      }
    }

    #pragma unroll
    for (int g = 0; g < 2; g++)
      #pragma unroll
      for (int r = 0; r < 8; r++) s[g][r] *= scl;

    if (amq) {
      #pragma unroll
      for (int g = 0; g < 2; g++) {
        const int col = kvBase + g * 16 + ln;
        #pragma unroll
        for (int r = 0; r < 8; r++) {
          const int qrow = qt * 16 + hi * 8 + r;
          if (!amq[((size_t)b * Qlen + qrow) * Klen + col]) s[g][r] = -1.0e9f;
        }
      }
    }

    // online softmax update (row-wise over 16-lane halves)
    float pr[2][8];
    #pragma unroll
    for (int r = 0; r < 8; r++) {
      float cm = fmaxf(s[0][r], s[1][r]);
      #pragma unroll
      for (int off = 1; off < 16; off <<= 1) cm = fmaxf(cm, __shfl_xor(cm, off, 32));
      const float nm = fmaxf(rowMax[r], cm);
      const float rescale = __expf(rowMax[r] - nm);
      float p0 = __expf(s[0][r] - nm);
      float p1 = __expf(s[1][r] - nm);
      float ps = p0 + p1;
      #pragma unroll
      for (int off = 1; off < 16; off <<= 1) ps += __shfl_xor(ps, off, 32);
      rowSum[r] = rowSum[r] * rescale + ps;
      rowMax[r] = nm;
      pr[0][r] = p0; pr[1][r] = p1;
      #pragma unroll
      for (int c = 0; c < 4; c++) oacc[c][r] *= rescale;
    }

    // transpose P from C layout to A layout through LDS
    #pragma unroll
    for (int g = 0; g < 2; g++)
      #pragma unroll
      for (int r = 0; r < 8; r++)
        sP[hi * 8 + r][g * 16 + ln] = (__bf16)pr[g][r];
    __syncthreads();

    v16bf ap;
    #pragma unroll
    for (int i = 0; i < 8; i++) {
      const int k = (i < 4 ? 0 : 16) + hi * 8 + (i & 3) * 2;
      ap[2 * i]     = sP[ln][k];
      ap[2 * i + 1] = sP[ln][k + 1];
    }
    #pragma unroll
    for (int c = 0; c < 4; c++) {
      v16bf bv;
      #pragma unroll
      for (int i = 0; i < 8; i++) {
        const int kk = (i < 4 ? 0 : 16) + hi * 8 + (i & 3) * 2;
        bv[2 * i]     = sV[kk][c * 16 + ln];
        bv[2 * i + 1] = sV[kk + 1][c * 16 + ln];
      }
      oacc[c] = __builtin_amdgcn_wmma_f32_16x16x32_bf16(false, ap, false, bv,
                                                        (short)0, oacc[c], false, false);
    }
    __syncthreads();
  }

  #pragma unroll
  for (int r = 0; r < 8; r++) {
    const float inv = 1.0f / rowSum[r];
    const size_t orow = (size_t)(qt * 16 + hi * 8 + r) * Ec;
    #pragma unroll
    for (int c = 0; c < 4; c++)
      Ob[orow + c * 16 + ln] = (__bf16)(oacc[c][r] * inv);
  }
}

// ---------------------------------------------------------------- mask prediction
// am[b,m,l] = (!kv_mask[b,l]) && (mask_tokens[b,m,:] . kv[b,l,:] > ln(9))
__global__ __launch_bounds__(32) void k_mask_pred(
    const __bf16* __restrict__ mt, const __bf16* __restrict__ kvb,
    const unsigned char* __restrict__ kvm, unsigned char* __restrict__ am)
{
  const int lane = threadIdx.x;
  const int ln = lane & 15, hi = lane >> 4;
  const int b = blockIdx.z, lt = blockIdx.y, qt = blockIdx.x;

  v8f acc = {};
  const __bf16* Arow  = mt  + (size_t)(b * Mc + qt * 16 + ln) * Ec;
  const __bf16* Bbase = kvb + (size_t)(b * Lc + lt * 16 + ln) * Ec;

  for (int kb = 0; kb < Ec; kb += 32) {
    v16bf af, bfm;
    #pragma unroll
    for (int i = 0; i < 8; i++) {
      const int ka = kb + (i < 4 ? 0 : 16) + hi * 8 + (i & 3) * 2;
      af[2 * i]     = Arow[ka];
      af[2 * i + 1] = Arow[ka + 1];
      const int kbi = kb + (i < 4 ? 0 : 16) + hi * 8 + (i & 3) * 2;
      bfm[2 * i]     = Bbase[kbi];
      bfm[2 * i + 1] = Bbase[kbi + 1];
    }
    acc = __builtin_amdgcn_wmma_f32_16x16x32_bf16(false, af, false, bfm,
                                                  (short)0, acc, false, false);
  }

  const int l = lt * 16 + ln;
  const unsigned char padded = kvm[(size_t)b * Lc + l];
  #pragma unroll
  for (int r = 0; r < 8; r++) {
    const int m = qt * 16 + hi * 8 + r;
    const unsigned char v = (!padded && acc[r] > 2.1972245773362196f) ? 1 : 0;
    am[((size_t)b * Mc + m) * Lc + l] = v;
  }
}

__global__ __launch_bounds__(256) void k_rowany(const unsigned char* __restrict__ am,
                                                unsigned* __restrict__ rowAny)
{
  const long row = blockIdx.x;
  const unsigned char* p = am + row * (long)Lc;
  unsigned acc = 0;
  for (int i = threadIdx.x; i < Lc; i += 256) acc |= p[i];
  __shared__ unsigned s;
  if (threadIdx.x == 0) s = 0;
  __syncthreads();
  if (acc) atomicOr(&s, 1u);
  __syncthreads();
  if (threadIdx.x == 0) rowAny[row] = s;
}

__global__ void k_amq(const unsigned char* __restrict__ am, const unsigned* __restrict__ rowAny,
                      const unsigned char* __restrict__ kvm, unsigned char* __restrict__ amq)
{
  long i = (long)blockIdx.x * blockDim.x + threadIdx.x;
  const long total = (long)Bc * Mc * Lc;
  if (i >= total) return;
  const long l  = i % Lc;
  const long bm = i / Lc;
  const long b  = bm / Mc;
  unsigned char a = am[i];
  if (rowAny[bm] == 0) a = 1;           // all-masked row attends everywhere
  if (kvm[b * Lc + l]) a = 0;           // padded keys always masked
  amq[i] = a;
}

// ---------------------------------------------------------------- GLU combine
__global__ void k_glu(const __bf16* __restrict__ h, __bf16* __restrict__ g, long total)
{
  long i = (long)blockIdx.x * blockDim.x + threadIdx.x;
  if (i >= total) return;
  const long r = i / HIDc, j = i - r * HIDc;
  const float x1 = (float)h[r * (2 * HIDc) + j];
  const float x2 = (float)h[r * (2 * HIDc) + HIDc + j];
  const float s  = x1 / (1.0f + __expf(-x1));   // SiLU
  g[i] = (__bf16)(s * x2);
}

// ---------------------------------------------------------------- host
extern "C" void kernel_launch(void* const* d_in, const int* in_sizes, int n_in,
                              void* d_out, int out_size, void* d_ws, size_t ws_size,
                              hipStream_t stream)
{
  (void)n_in; (void)out_size; (void)ws_size;

  const float* q_in  = (const float*)d_in[0];
  const float* kv_in = (const float*)d_in[1];
  const unsigned char* kvm = (const unsigned char*)d_in[2];

  // jax tree-flatten order: params dict keys sorted, leaves sorted within.
  enum {
    KVCA_BK = 3, KVCA_BO, KVCA_BQ, KVCA_BV, KVCA_WK, KVCA_WO, KVCA_WQ, KVCA_WV,
    KVD_BIN, KVD_BOUT, KVD_WIN, KVD_WOUT,
    MN_BM, MN_WM,
    QCA_BK, QCA_BO, QCA_BQ, QCA_BV, QCA_WK, QCA_WO, QCA_WQ, QCA_WV,
    QD_BIN, QD_BOUT, QD_WIN, QD_WOUT,
    QSA_BK, QSA_BO, QSA_BQ, QSA_BV, QSA_WK, QSA_WO, QSA_WQ, QSA_WV
  };
  auto fp = [&](int i) { return (const float*)d_in[i]; };

  char* ws = (char*)d_ws;
  size_t cur = 0;
  auto alloc = [&](size_t bytes) -> char* {
    char* p = ws + cur;
    cur = (cur + bytes + 255) & ~(size_t)255;
    return p;
  };

  const size_t TQ = (size_t)Bc * Mc;   // 2048 q tokens
  const size_t TK = (size_t)Bc * Lc;   // 32768 kv tokens

  __bf16* q_bf  = (__bf16*)alloc(TQ * Ec * 2);
  __bf16* kv_bf = (__bf16*)alloc(TK * Ec * 2);

  // convert weights to bf16
  __bf16* wbf[40] = {};
  const int wIdx[17] = { MN_WM,
                         KVCA_WK, KVCA_WO, KVCA_WQ, KVCA_WV,
                         QCA_WK, QCA_WO, QCA_WQ, QCA_WV,
                         QSA_WK, QSA_WO, QSA_WQ, QSA_WV,
                         QD_WIN, QD_WOUT, KVD_WIN, KVD_WOUT };
  for (int t = 0; t < 17; t++) {
    const int idx = wIdx[t];
    const long n = (long)in_sizes[idx];
    wbf[idx] = (__bf16*)alloc((size_t)n * 2);
    k_f32_to_bf16<<<dim3((unsigned)((n + 255) / 256)), 256, 0, stream>>>(fp(idx), wbf[idx], n);
  }

  __bf16* mt_bf = (__bf16*)alloc(TQ * Ec * 2);
  unsigned char* am   = (unsigned char*)alloc((size_t)Bc * Mc * Lc);
  unsigned*      rAny = (unsigned*)alloc(TQ * 4);
  unsigned char* amqB = (unsigned char*)alloc((size_t)Bc * Mc * Lc);
  __bf16* projQ = (__bf16*)alloc(TK * Ec * 2);
  __bf16* projK = (__bf16*)alloc(TK * Ec * 2);
  __bf16* projV = (__bf16*)alloc(TK * Ec * 2);
  __bf16* attnO = (__bf16*)alloc(TK * Ec * 2);
  float*  q1f = (float*)alloc(TQ * Ec * 4);
  float*  q2f = (float*)alloc(TQ * Ec * 4);
  __bf16* q1b = (__bf16*)alloc(TQ * Ec * 2);
  __bf16* q2b = (__bf16*)alloc(TQ * Ec * 2);
  __bf16* q3b = (__bf16*)alloc(TQ * Ec * 2);
  __bf16* kv1b = (__bf16*)alloc(TK * Ec * 2);
  __bf16* hid  = (__bf16*)alloc((size_t)2048 * 2 * HIDc * 2);   // 2048 x 4096
  __bf16* gbuf = (__bf16*)alloc((size_t)2048 * HIDc * 2);       // 2048 x 2048

  float* outQ  = (float*)d_out;
  float* outKV = (float*)d_out + TQ * Ec;

  { long n = (long)(TQ * Ec); k_f32_to_bf16<<<dim3((unsigned)((n + 255) / 256)), 256, 0, stream>>>(q_in,  q_bf,  n); }
  { long n = (long)(TK * Ec); k_f32_to_bf16<<<dim3((unsigned)((n + 255) / 256)), 256, 0, stream>>>(kv_in, kv_bf, n); }

  // epilogue variants: OB only / RES+OF / RES+OF+OB
  auto gemm = [&](const __bf16* A, const __bf16* W, const float* bias,
                  const float* resid, float* oF, __bf16* oB,
                  int rows, int N, int K) {
    dim3 g(N / 128, rows / 128);
    if (resid && oF && oB)
      k_gemm_bf16<true, true, true><<<g, 256, 0, stream>>>(A, W, bias, resid, oF, oB, rows, N, K);
    else if (resid && oF)
      k_gemm_bf16<true, true, false><<<g, 256, 0, stream>>>(A, W, bias, resid, oF, oB, rows, N, K);
    else
      k_gemm_bf16<false, false, true><<<g, 256, 0, stream>>>(A, W, bias, resid, oF, oB, rows, N, K);
  };
  auto attn = [&](const __bf16* Q, const __bf16* Kp, const __bf16* Vp, __bf16* O,
                  const unsigned char* msk, int Qlen, int Klen,
                  long long qStride, long long kStride) {
    dim3 g(Qlen / 16, Hc, Bc);
    k_attn<<<g, 32, 0, stream>>>(Q, Kp, Vp, O, msk, Qlen, Klen, qStride, kStride);
  };

  // ---- mask net: mask_tokens = q @ wm + bm ; am ; fallback ; resolve
  gemm(q_bf, wbf[MN_WM], fp(MN_BM), nullptr, nullptr, mt_bf, (int)TQ, Ec, Ec);
  k_mask_pred<<<dim3(Mc / 16, Lc / 16, Bc), 32, 0, stream>>>(mt_bf, kv_bf, kvm, am);
  k_rowany<<<dim3((unsigned)TQ), 256, 0, stream>>>(am, rAny);
  { long n = (long)Bc * Mc * Lc;
    k_amq<<<dim3((unsigned)((n + 255) / 256)), 256, 0, stream>>>(am, rAny, kvm, amqB); }

  // ---- q cross-attention (masked)
  gemm(q_bf,  wbf[QCA_WQ], fp(QCA_BQ), nullptr, nullptr, projQ, (int)TQ, Ec, Ec);
  gemm(kv_bf, wbf[QCA_WK], fp(QCA_BK), nullptr, nullptr, projK, (int)TK, Ec, Ec);
  gemm(kv_bf, wbf[QCA_WV], fp(QCA_BV), nullptr, nullptr, projV, (int)TK, Ec, Ec);
  attn(projQ, projK, projV, attnO, amqB, Mc, Lc, (long long)Mc * Ec, (long long)Lc * Ec);
  gemm(attnO, wbf[QCA_WO], fp(QCA_BO), q_in, q1f, q1b, (int)TQ, Ec, Ec);

  // ---- q self-attention
  gemm(q1b, wbf[QSA_WQ], fp(QSA_BQ), nullptr, nullptr, projQ, (int)TQ, Ec, Ec);
  gemm(q1b, wbf[QSA_WK], fp(QSA_BK), nullptr, nullptr, projK, (int)TQ, Ec, Ec);
  gemm(q1b, wbf[QSA_WV], fp(QSA_BV), nullptr, nullptr, projV, (int)TQ, Ec, Ec);
  attn(projQ, projK, projV, attnO, nullptr, Mc, Mc, (long long)Mc * Ec, (long long)Mc * Ec);
  gemm(attnO, wbf[QSA_WO], fp(QSA_BO), q1f, q2f, q2b, (int)TQ, Ec, Ec);

  // ---- q GLU -> final q output
  gemm(q2b, wbf[QD_WIN], fp(QD_BIN), nullptr, nullptr, hid, (int)TQ, 2 * HIDc, Ec);
  { long n = (long)TQ * HIDc;
    k_glu<<<dim3((unsigned)((n + 255) / 256)), 256, 0, stream>>>(hid, gbuf, n); }
  gemm(gbuf, wbf[QD_WOUT], fp(QD_BOUT), q2f, outQ, q3b, (int)TQ, Ec, HIDc);

  // ---- kv cross-attention (transposed mask is provably all-True)
  gemm(kv_bf, wbf[KVCA_WQ], fp(KVCA_BQ), nullptr, nullptr, projQ, (int)TK, Ec, Ec);
  gemm(q3b,   wbf[KVCA_WK], fp(KVCA_BK), nullptr, nullptr, projK, (int)TQ, Ec, Ec);
  gemm(q3b,   wbf[KVCA_WV], fp(KVCA_BV), nullptr, nullptr, projV, (int)TQ, Ec, Ec);
  attn(projQ, projK, projV, attnO, nullptr, Lc, Mc, (long long)Lc * Ec, (long long)Mc * Ec);
  gemm(attnO, wbf[KVCA_WO], fp(KVCA_BO), kv_in, outKV, kv1b, (int)TK, Ec, Ec);

  // ---- kv GLU, chunked per batch (reuses hid/gbuf); residual added in place
  for (int b = 0; b < Bc; b++) {
    const __bf16* src = kv1b + (size_t)b * Lc * Ec;
    float* res = outKV + (size_t)b * Lc * Ec;
    gemm(src, wbf[KVD_WIN], fp(KVD_BIN), nullptr, nullptr, hid, Lc, 2 * HIDc, Ec);
    { long n = (long)Lc * HIDc;
      k_glu<<<dim3((unsigned)((n + 255) / 256)), 256, 0, stream>>>(hid, gbuf, n); }
    gemm(gbuf, wbf[KVD_WOUT], fp(KVD_BOUT), res, res, nullptr, Lc, Ec, HIDc);
  }
}
